// GCN_LSTM_16166256902762
// MI455X (gfx1250) — matrix operand order repaired and checked by reading, not compiled, not run on hardware
//
#include <hip/hip_runtime.h>
#include <math.h>

typedef __bf16 bf16_t;
typedef __attribute__((ext_vector_type(16))) __bf16 v16bf;
typedef __attribute__((ext_vector_type(8)))  __bf16 v8bf;
typedef __attribute__((ext_vector_type(4)))  __bf16 v4bf;
typedef __attribute__((ext_vector_type(8)))  float  v8f;

// ---------------------------------------------------------------------------
// WMMA GEMM:  H[nrows, fo] = Xbf[nrows, fi] @ W  (both sides bf16, f32 accum)
// one wave -> one 16x16 tile, K stepped by 32.
// A: row-major bf16; per-lane fragment = two contiguous 16B loads.
// B: pre-packed fragments; frag(nt,kt) at ((nt*ktiles+kt)*32 + lane)*16 + j.
// ---------------------------------------------------------------------------
__global__ void __launch_bounds__(128)
k_gemm_wmma(const bf16_t* __restrict__ Xbf, const bf16_t* __restrict__ Bp,
            float* __restrict__ H, int nrows, int fi, int fo) {
  const int lane  = threadIdx.x & 31;
  const int wid   = threadIdx.x >> 5;
  const int ntile = fo >> 4;
  const int nt    = blockIdx.y * 4 + wid;
  if (nt >= ntile) return;                      // wave-uniform exit
  const int mt   = blockIdx.x;
  const int half = lane >> 4;                   // 0: K 0..7/16..23, 1: K 8..15/24..31
  const int l16  = lane & 15;
  int row = mt * 16 + l16;
  if (row >= nrows) row = nrows - 1;            // clamp (N=50000 is exact anyway)
  const int ktiles = fi >> 5;

  const bf16_t* xrow  = Xbf + (size_t)row * fi + half * 8;
  const bf16_t* bbase = Bp + ((size_t)nt * ktiles * 32 + lane) * 16;

  v8f acc = {};
  for (int kt = 0; kt < ktiles; ++kt) {
    v8bf a0 = *reinterpret_cast<const v8bf*>(xrow + kt * 32);
    v8bf a1 = *reinterpret_cast<const v8bf*>(xrow + kt * 32 + 16);
    v16bf a;
#pragma unroll
    for (int j = 0; j < 8; ++j) { a[j] = a0[j]; a[8 + j] = a1[j]; }
    v16bf b = *reinterpret_cast<const v16bf*>(bbase + (size_t)kt * 512);
    acc = __builtin_amdgcn_wmma_f32_16x16x32_bf16(false, a, false, b,
                                                  (short)0, acc, false, false);
  }
  const int col = nt * 16 + l16;
  float* hp = H + (size_t)mt * 16 * fo + col;
#pragma unroll
  for (int r = 0; r < 8; ++r) {
    int rr = mt * 16 + half * 8 + r;
    if (rr < nrows) hp[(size_t)(half * 8 + r) * fo] = acc[r];
  }
}

// f32 -> bf16 conversion pass (4 elements / thread, 8B stores)
__global__ void k_cvt_bf16(const float* __restrict__ in, bf16_t* __restrict__ out,
                           long long n4) {
  long long t = (long long)blockIdx.x * blockDim.x + threadIdx.x;
  if (t >= n4) return;
  float4 f = reinterpret_cast<const float4*>(in)[t];
  v4bf v;
  v[0] = (bf16_t)f.x; v[1] = (bf16_t)f.y; v[2] = (bf16_t)f.z; v[3] = (bf16_t)f.w;
  *reinterpret_cast<v4bf*>(out + 4 * t) = v;
}

// Pack f32 W[fi][fo] (row-major, k-major) -> bf16 WMMA-B fragments.
__global__ void k_pack_w(const float* __restrict__ W, bf16_t* __restrict__ Bp,
                         int fi, int fo) {
  int t = blockIdx.x * blockDim.x + threadIdx.x;
  if (t >= fi * fo) return;
  int j      = t & 15;
  int lane   = (t >> 4) & 31;
  int rest   = t >> 9;
  int ktiles = fi >> 5;
  int kt     = rest % ktiles;
  int nt     = rest / ktiles;
  int k      = kt * 32 + ((lane & 16) ? 8 : 0) + (j & 7) + ((j & 8) ? 16 : 0);
  int n      = nt * 16 + (lane & 15);
  Bp[t] = (bf16_t)W[(size_t)k * fo + n];
}

// Small scalar GEMM for conv1 (3->64) and conv2 (64->3).
__global__ void k_gemm_small(const float* __restrict__ X, const float* __restrict__ W,
                             float* __restrict__ H, int n, int fi, int fo) {
  int t = blockIdx.x * blockDim.x + threadIdx.x;
  if (t >= n * fo) return;
  int row = t / fo, col = t % fo;
  float acc = 0.f;
  for (int k = 0; k < fi; ++k) acc += X[(size_t)row * fi + k] * W[(size_t)k * fo + col];
  H[t] = acc;
}

// ---------------------------------------------------------------------------
// GCN aggregation: OUT = b + H*dis^2 (self term) + scatter_add(H[src]*w_e, dst)
// ---------------------------------------------------------------------------
__global__ void k_agg_init(const float* __restrict__ H, const float* __restrict__ dis,
                           const float* __restrict__ bias, float* __restrict__ OUT,
                           long long total, int fo) {
  long long t = (long long)blockIdx.x * blockDim.x + threadIdx.x;
  if (t >= total) return;
  int row = (int)(t / fo), c = (int)(t % fo);
  float d = dis[row];
  OUT[t] = H[t] * d * d + bias[c];
}

__global__ void k_edge_scatter4(const int* __restrict__ src, const int* __restrict__ dst,
                                const float* __restrict__ dis, const float* __restrict__ H,
                                float* __restrict__ OUT, long long total, int fo) {
  long long t = (long long)blockIdx.x * blockDim.x + threadIdx.x;
  if (t >= total) return;
  int groups = fo >> 2;
  int e  = (int)(t / groups);
  int gi = (int)(t % groups);
  int s = src[e], d = dst[e];
  float w = dis[s] * dis[d];
  float4 h = *reinterpret_cast<const float4*>(H + (size_t)s * fo + gi * 4);
  float* ap = OUT + (size_t)d * fo + gi * 4;
  atomicAdd(ap + 0, h.x * w);
  atomicAdd(ap + 1, h.y * w);
  atomicAdd(ap + 2, h.z * w);
  atomicAdd(ap + 3, h.w * w);
}

__global__ void k_edge_scatter1(const int* __restrict__ src, const int* __restrict__ dst,
                                const float* __restrict__ dis, const float* __restrict__ H,
                                float* __restrict__ OUT, long long total, int fo) {
  long long t = (long long)blockIdx.x * blockDim.x + threadIdx.x;
  if (t >= total) return;
  int e = (int)(t / fo), c = (int)(t % fo);
  int s = src[e], d = dst[e];
  float w = dis[s] * dis[d];
  atomicAdd(OUT + (size_t)d * fo + c, H[(size_t)s * fo + c] * w);
}

// ---------------------------------------------------------------------------
// degree / dis = rsqrt(1 + indeg)
// ---------------------------------------------------------------------------
__global__ void k_fill(float* p, float v, long long n) {
  long long t = (long long)blockIdx.x * blockDim.x + threadIdx.x;
  if (t < n) p[t] = v;
}
__global__ void k_deg(const int* __restrict__ dst, float* __restrict__ deg, int E) {
  int t = blockIdx.x * blockDim.x + threadIdx.x;
  if (t < E) atomicAdd(&deg[dst[t]], 1.0f);
}
__global__ void k_rsqrt(float* p, int n) {
  int t = blockIdx.x * blockDim.x + threadIdx.x;
  if (t < n) p[t] = rsqrtf(p[t]);
}

// ---------------------------------------------------------------------------
// global column max over all rows (== global_max_pool then max over graphs)
// ---------------------------------------------------------------------------
__global__ void k_colmax(const float* __restrict__ X, float* __restrict__ g,
                         int n, int fo) {
  int col   = threadIdx.x;                // blockDim.x == fo (256)
  int chunk = (n + gridDim.x - 1) / gridDim.x;
  int r0 = blockIdx.x * chunk;
  int r1 = r0 + chunk; if (r1 > n) r1 = n;
  float m = -3.0e38f;
  for (int r = r0; r < r1; ++r) m = fmaxf(m, X[(size_t)r * fo + col]);
  unsigned* gp = reinterpret_cast<unsigned*>(g + col);
  float old = __uint_as_float(*gp);
  while (m > old) {
    unsigned assumed = __float_as_uint(old);
    unsigned prev = atomicCAS(gp, assumed, __float_as_uint(m));
    if (prev == assumed) break;
    old = __uint_as_float(prev);
  }
}

// cat = [ broadcast g (256) | t2 (256) | xin (64) ] -> bf16 [N, 576]
// (the concat is only ever consumed by the c7 WMMA GEMM -> emit bf16 directly)
__global__ void k_build_cat_bf16(const float* __restrict__ g, const float* __restrict__ t2,
                                 const float* __restrict__ xin, bf16_t* __restrict__ C,
                                 long long total) {
  long long t = (long long)blockIdx.x * blockDim.x + threadIdx.x;
  if (t >= total) return;
  int row = (int)(t / 576), c = (int)(t % 576);
  float v;
  if (c < 256)      v = g[c];
  else if (c < 512) v = t2[(size_t)row * 256 + (c - 256)];
  else              v = xin[(size_t)row * 64 + (c - 512)];
  C[t] = (bf16_t)v;
}

// ---------------------------------------------------------------------------
// elementwise glue
// ---------------------------------------------------------------------------
__global__ void k_unary(const float* __restrict__ in, float* __restrict__ out,
                        long long n, int mode) {
  long long t = (long long)blockIdx.x * blockDim.x + threadIdx.x;
  if (t >= n) return;
  float v = in[t];
  if (mode == 1) v = tanhf(v);
  else if (mode == 2) v = 1.0f / (1.0f + expf(-v));
  out[t] = v;
}
__global__ void k_mul(const float* a, const float* b, float* o, long long n) {
  long long t = (long long)blockIdx.x * blockDim.x + threadIdx.x;
  if (t < n) o[t] = a[t] * b[t];
}
__global__ void k_fma(const float* a, const float* b, const float* c, float* o,
                      long long n) {
  long long t = (long long)blockIdx.x * blockDim.x + threadIdx.x;
  if (t < n) o[t] = a[t] * b[t] + c[t];
}

// ---------------------------------------------------------------------------
// host-side orchestration
// ---------------------------------------------------------------------------
namespace {

inline int divll(long long a, int b) { return (int)((a + b - 1) / b); }

struct Net {
  hipStream_t st;
  int N, E;
  const int *src, *dst;
  const float* dis;
  bf16_t* Xbf;      // shared bf16 staging buffer, N*576 elems
};

// full GCN layer with WMMA GEMM; OUT may alias Xf.
// If Xf != nullptr, converts Xf -> Xbf first; otherwise Xbf must already hold A.
void gcn_wmma(const Net& nt, const float* Xf, const bf16_t* Wp, const float* bias,
              float* H, float* OUT, int fi, int fo, int act) {
  if (Xf) {
    long long n4 = (long long)nt.N * fi / 4;
    k_cvt_bf16<<<divll(n4, 256), 256, 0, nt.st>>>(Xf, nt.Xbf, n4);
  }
  dim3 gg(nt.N / 16 + ((nt.N % 16) ? 1 : 0), (fo / 16 + 3) / 4);
  k_gemm_wmma<<<gg, 128, 0, nt.st>>>(nt.Xbf, Wp, H, nt.N, fi, fo);
  long long nf = (long long)nt.N * fo;
  k_agg_init<<<divll(nf, 256), 256, 0, nt.st>>>(H, nt.dis, bias, OUT, nf, fo);
  long long te = (long long)nt.E * (fo >> 2);
  k_edge_scatter4<<<divll(te, 256), 256, 0, nt.st>>>(nt.src, nt.dst, nt.dis, H, OUT, te, fo);
  if (act) k_unary<<<divll(nf, 256), 256, 0, nt.st>>>(OUT, OUT, nf, act);
}

void inception(const Net& nt, const float* in, bf16_t* const Wp[5],
               const float* const Wb[5], float* t1, float* t2,
               float* H, float* g, float* out) {
  gcn_wmma(nt, in, Wp[0], Wb[0], H, t1, 64, 128, 1);    // c1 + tanh
  gcn_wmma(nt, t1, Wp[1], Wb[1], H, t1, 128, 256, 1);   // c2 + tanh
  k_fill<<<1, 256, 0, nt.st>>>(g, -3.0e38f, 256);
  k_colmax<<<64, 256, 0, nt.st>>>(t1, g, nt.N, 256);
  gcn_wmma(nt, in, Wp[2], Wb[2], H, t2, 64, 128, 1);    // c3 + tanh
  gcn_wmma(nt, t2, Wp[3], Wb[3], H, t2, 128, 256, 1);   // c4 + tanh
  long long tc = (long long)nt.N * 576;
  k_build_cat_bf16<<<divll(tc, 256), 256, 0, nt.st>>>(g, t2, in, nt.Xbf, tc);
  gcn_wmma(nt, nullptr, Wp[4], Wb[4], H, out, 576, 64, 1);  // c7 + tanh
}

} // namespace

extern "C" void kernel_launch(void* const* d_in, const int* in_sizes, int n_in,
                              void* d_out, int out_size, void* d_ws, size_t ws_size,
                              hipStream_t stream) {
  const int N = in_sizes[0] / 3;
  const int E = in_sizes[1] / 2;

  const float* x    = (const float*)d_in[0];
  const int*   srcI = (const int*)d_in[1];
  const int*   dstI = srcI + E;
  // d_in[2] = batch (unused by the reference math)
  const float* c1W = (const float*)d_in[3];
  const float* c1b = (const float*)d_in[4];
  const float* c2W = (const float*)d_in[5];
  const float* c2b = (const float*)d_in[6];
  const float* incW[6][5]; const float* incB[6][5];
  for (int i = 0; i < 6; ++i)
    for (int j = 0; j < 5; ++j) {
      incW[i][j] = (const float*)d_in[7 + i * 10 + 2 * j];
      incB[i][j] = (const float*)d_in[7 + i * 10 + 2 * j + 1];
    }
  static const int FI[5] = {64, 128, 64, 128, 576};
  static const int FO[5] = {128, 256, 128, 256, 64};

  // workspace carve (256B aligned chunks)
  char* w = (char*)d_ws;
  auto carve = [&](size_t bytes) -> char* {
    char* p = w; w += (bytes + 255) & ~(size_t)255; return p;
  };
  float*  dis  = (float*)carve((size_t)N * 4);
  float*  x0   = (float*)carve((size_t)N * 64 * 4);
  float*  fbuf = (float*)carve((size_t)N * 64 * 4);
  float*  i1b  = (float*)carve((size_t)N * 64 * 4);
  float*  i2b  = (float*)carve((size_t)N * 64 * 4);
  float*  t1   = (float*)carve((size_t)N * 256 * 4);
  float*  t2   = (float*)carve((size_t)N * 256 * 4);
  float*  H    = (float*)carve((size_t)N * 256 * 4);
  bf16_t* Xbf  = (bf16_t*)carve((size_t)N * 576 * 2);
  float*  gbuf = (float*)carve(1024 * 4);
  bf16_t* Wp[6][5];
  for (int i = 0; i < 6; ++i)
    for (int j = 0; j < 5; ++j)
      Wp[i][j] = (bf16_t*)carve((size_t)FI[j] * FO[j] * 2);

  Net nt{stream, N, E, srcI, dstI, dis, Xbf};

  // pack all inception weights into WMMA-B bf16 fragments (re-done every call)
  for (int i = 0; i < 6; ++i)
    for (int j = 0; j < 5; ++j) {
      int total = FI[j] * FO[j];
      k_pack_w<<<divll(total, 256), 256, 0, stream>>>(incW[i][j], Wp[i][j], FI[j], FO[j]);
    }

  // dis = rsqrt(1 + indegree)
  k_fill<<<divll(N, 256), 256, 0, stream>>>(dis, 1.0f, N);
  k_deg<<<divll(E, 256), 256, 0, stream>>>(dstI, dis, E);
  k_rsqrt<<<divll(N, 256), 256, 0, stream>>>(dis, N);

  // x0 = gcn(x, conv1)   (3 -> 64, no activation)
  k_gemm_small<<<divll((long long)N * 64, 256), 256, 0, stream>>>(x, c1W, H, N, 3, 64);
  {
    long long nf = (long long)N * 64;
    k_agg_init<<<divll(nf, 256), 256, 0, stream>>>(H, dis, c1b, x0, nf, 64);
    long long te = (long long)E * 16;
    k_edge_scatter4<<<divll(te, 256), 256, 0, stream>>>(srcI, dstI, dis, H, x0, te, 64);
  }

  long long n64 = (long long)N * 64;

  // f = inception(inception(x0, inc1), inc2);  f = x0 * f
  inception(nt, x0, Wp[0], incB[0], t1, t2, H, gbuf, fbuf);
  inception(nt, fbuf, Wp[1], incB[1], t1, t2, H, gbuf, fbuf);
  k_mul<<<divll(n64, 256), 256, 0, stream>>>(fbuf, x0, fbuf, n64);

  // i1 = sigmoid(inception(inception(x0, inc3), inc4))
  inception(nt, x0, Wp[2], incB[2], t1, t2, H, gbuf, i1b);
  inception(nt, i1b, Wp[3], incB[3], t1, t2, H, gbuf, i1b);
  k_unary<<<divll(n64, 256), 256, 0, stream>>>(i1b, i1b, n64, 2);

  // i2 = tanh(inception(inception(x0, inc5), inc6))
  inception(nt, x0, Wp[4], incB[4], t1, t2, H, gbuf, i2b);
  inception(nt, i2b, Wp[5], incB[5], t1, t2, H, gbuf, i2b);
  k_unary<<<divll(n64, 256), 256, 0, stream>>>(i2b, i2b, n64, 1);

  // i = i1*i2 + f   (into i1b)
  k_fma<<<divll(n64, 256), 256, 0, stream>>>(i1b, i2b, fbuf, i1b, n64);

  // out = tanh(gcn(i, conv2))   (64 -> 3), aggregated directly into d_out
  float* out = (float*)d_out;
  k_gemm_small<<<divll((long long)N * 3, 256), 256, 0, stream>>>(i1b, c2W, H, N, 64, 3);
  {
    long long nf = (long long)N * 3;
    k_agg_init<<<divll(nf, 256), 256, 0, stream>>>(H, dis, c2b, out, nf, 3);
    long long te = (long long)E * 3;
    k_edge_scatter1<<<divll(te, 256), 256, 0, stream>>>(srcI, dstI, dis, H, out, te, 3);
    k_unary<<<divll(nf, 256), 256, 0, stream>>>(out, out, nf, 1);
  }
}
